// CombinedModel_10264971837885
// MI455X (gfx1250) — matrix operand order repaired and checked by reading, not compiled, not run on hardware
//
#include <hip/hip_runtime.h>
#include <hip/hip_bf16.h>

// ---------------------------------------------------------------------------
// Model dims
// ---------------------------------------------------------------------------
#define BSZ     4
#define LSEQ    2048
#define DMODEL  2048
#define DINNER  4096
#define DSTATE  64
#define NHEADS  64
#define HEADDIM 64
#define CONVDIM 4224            // DINNER + 2*DSTATE
#define DINPROJ 8384            // 2*DINNER + 2*DSTATE + NHEADS
#define MROWS   (BSZ * LSEQ)    // 8192
#define NCLS    4

// ---------------------------------------------------------------------------
// WMMA types (gfx1250, wave32)
// ---------------------------------------------------------------------------
typedef __attribute__((ext_vector_type(16))) __bf16        v16bf;
typedef __attribute__((ext_vector_type(8)))  float         v8f;
typedef __attribute__((ext_vector_type(4)))  unsigned int  u32x4;

union FragBF { v16bf bf; u32x4 u[2]; };

__device__ __forceinline__ unsigned short f32_to_bf16_rne(float f) {
    unsigned int u = __float_as_uint(f);
    unsigned int r = u + 0x7FFFu + ((u >> 16) & 1u);
    return (unsigned short)(r >> 16);
}

// ---------------------------------------------------------------------------
// f32 -> bf16 conversion (grid-stride)
// ---------------------------------------------------------------------------
__global__ __launch_bounds__(256)
void k_f32_to_bf16(const float* __restrict__ in, unsigned short* __restrict__ out,
                   size_t n) {
    size_t i = (size_t)blockIdx.x * blockDim.x + threadIdx.x;
    size_t stride = (size_t)gridDim.x * blockDim.x;
    for (; i < n; i += stride) out[i] = f32_to_bf16_rne(in[i]);
}

// ---------------------------------------------------------------------------
// Register-blocked WMMA GEMM:  C[M,N] = A[M,K] * B[N,K]^T
// (A,B bf16 row-major; C f32)
//
// Block = 256 threads = 8 waves stacked along M; block tile 512(M) x 64(N).
// Each wave computes a 64x64 macro-tile = 4x4 grid of 16x16 WMMA tiles
// (16 x v8f accumulators). Per K-step of 32: 4 A-frags + 4 B-frags feed
// 16 independent v_wmma_f32_16x16x32_bf16 -> 32 FLOP/byte register reuse.
// All 8 waves share identical B fragments (L0/L2 hits).
// Requires M % 512 == 0, N % 64 == 0, K % 32 == 0 (true for all call sites).
// ---------------------------------------------------------------------------
__global__ __launch_bounds__(256)
void k_wmma_gemm_bf16(const unsigned short* __restrict__ A,
                      const unsigned short* __restrict__ B,
                      float* __restrict__ C,
                      int M, int N, int K) {
    const int lane = threadIdx.x & 31;
    const int wave = threadIdx.x >> 5;
    const int half = lane >> 4;     // K-subset selector
    const int r    = lane & 15;     // A row / B col within 16x16 tile

    const int tile_m = blockIdx.y * 512 + wave * 64;  // wave's 64-row stripe
    const int tile_n = blockIdx.x * 64;               // block's 64-col stripe

    const unsigned short* Arow[4];
    const unsigned short* Brow[4];
#pragma unroll
    for (int i = 0; i < 4; ++i) {
        Arow[i] = A + (size_t)(tile_m + i * 16 + r) * K;
        Brow[i] = B + (size_t)(tile_n + i * 16 + r) * K;
    }

    v8f acc[4][4] = {};

    for (int k0 = 0; k0 < K; k0 += 32) {
        FragBF a[4], b[4];
#pragma unroll
        for (int i = 0; i < 4; ++i) {
            // prefetch ~16 iterations ahead (speculative, OOB-safe)
            __builtin_prefetch(Arow[i] + k0 + 512, 0, 1);
            __builtin_prefetch(Brow[i] + k0 + 512, 0, 1);
            // A 16x32: lanes 0-15 hold K {0..7,16..23}; lanes 16-31 {8..15,24..31}
            a[i].u[0] = *(const u32x4*)(Arow[i] + k0 + half * 8);
            a[i].u[1] = *(const u32x4*)(Arow[i] + k0 + 16 + half * 8);
            // B 32x16: lanes 0-15 hold K {0..7,16..23}; lanes 16-31 {8..15,24..31}
            b[i].u[0] = *(const u32x4*)(Brow[i] + k0 + half * 8);
            b[i].u[1] = *(const u32x4*)(Brow[i] + k0 + 16 + half * 8);
        }
#pragma unroll
        for (int i = 0; i < 4; ++i)
#pragma unroll
            for (int j = 0; j < 4; ++j)
                acc[i][j] = __builtin_amdgcn_wmma_f32_16x16x32_bf16(
                    /*neg_a=*/false, a[i].bf, /*neg_b=*/false, b[j].bf,
                    /*c_mod=*/(short)0, acc[i][j],
                    /*reuse_a=*/false, /*reuse_b=*/false);
    }

    // C/D layout: VGPR v holds row M = half*8 + v, col N = r
#pragma unroll
    for (int i = 0; i < 4; ++i)
#pragma unroll
        for (int j = 0; j < 4; ++j) {
            float* Crow = C + (size_t)(tile_m + i * 16 + half * 8) * N
                            + (tile_n + j * 16 + r);
#pragma unroll
            for (int v = 0; v < 8; ++v) Crow[(size_t)v * N] = acc[i][j][v];
        }
}

// ---------------------------------------------------------------------------
// Depthwise causal conv (D_CONV=4) over xBC slice of zxbcdt, + SiLU.
// One thread per (row, channel); rows = b*LSEQ + l.
// ---------------------------------------------------------------------------
__global__ __launch_bounds__(256)
void k_conv_silu(const float* __restrict__ z0,        // [MROWS, DINPROJ]
                 const float* __restrict__ conv_w,    // [CONVDIM, 4]
                 const float* __restrict__ conv_b,    // [CONVDIM]
                 float* __restrict__ xbc) {           // [MROWS, CONVDIM]
    size_t idx = (size_t)blockIdx.x * blockDim.x + threadIdx.x;
    const size_t total = (size_t)MROWS * CONVDIM;
    if (idx >= total) return;
    int c   = (int)(idx % CONVDIM);
    int row = (int)(idx / CONVDIM);
    int l   = row & (LSEQ - 1);
    float acc = conv_b[c];
#pragma unroll
    for (int k = 0; k < 4; ++k) {
        int lt = l - 3 + k;
        if (lt >= 0)
            acc += z0[(size_t)(row - 3 + k) * DINPROJ + DINNER + c] * conv_w[c * 4 + k];
    }
    xbc[idx] = acc / (1.f + __expf(-acc));   // SiLU
}

// ---------------------------------------------------------------------------
// dt = softplus(dt_raw + dt_bias);  dA = exp(-exp(A_log) * dt)
// ---------------------------------------------------------------------------
__global__ __launch_bounds__(256)
void k_dt(const float* __restrict__ z0, const float* __restrict__ dt_bias,
          const float* __restrict__ A_log,
          float* __restrict__ dtv, float* __restrict__ dAv) {
    int idx = blockIdx.x * blockDim.x + threadIdx.x;   // MROWS*NHEADS
    if (idx >= MROWS * NHEADS) return;
    int h   = idx & (NHEADS - 1);
    int row = idx >> 6;
    float x  = z0[(size_t)row * DINPROJ + (DINNER + CONVDIM) + h] + dt_bias[h];
    float dt = (x > 20.f) ? x : log1pf(__expf(x));
    dtv[idx] = dt;
    dAv[idx] = __expf(-__expf(A_log[h]) * dt);
}

// ---------------------------------------------------------------------------
// Sequential SSM scan. Grid (NHEADS, BSZ), block 64 (= headdim lanes).
// Each lane keeps its 64-entry state row in registers; B/C staged via LDS.
// ---------------------------------------------------------------------------
__global__ __launch_bounds__(64)
void k_ssm_scan(const float* __restrict__ xbc,   // [MROWS, CONVDIM]
                const float* __restrict__ dtv,   // [MROWS, NHEADS]
                const float* __restrict__ dAv,   // [MROWS, NHEADS]
                const float* __restrict__ Dp,    // [NHEADS]
                float* __restrict__ y_ssm) {     // [MROWS, DINNER]
    const int h = blockIdx.x;
    const int b = blockIdx.y;
    const int p = threadIdx.x;
    const float Dh = Dp[h];

    float st[DSTATE];
#pragma unroll
    for (int n = 0; n < DSTATE; ++n) st[n] = 0.f;

    __shared__ float Bs[DSTATE];
    __shared__ float Cs[DSTATE];

    for (int t = 0; t < LSEQ; ++t) {
        const size_t row = (size_t)b * LSEQ + t;
        const float* xr = xbc + row * CONVDIM;
        Bs[p] = xr[DINNER + p];
        Cs[p] = xr[DINNER + DSTATE + p];
        __syncthreads();

        float dA  = dAv[row * NHEADS + h];
        float dt  = dtv[row * NHEADS + h];
        float xh  = xr[h * HEADDIM + p];
        float dtx = dt * xh;

        float y = 0.f;
#pragma unroll
        for (int n = 0; n < DSTATE; ++n) {
            st[n] = st[n] * dA + dtx * Bs[n];
            y += st[n] * Cs[n];
        }
        y_ssm[row * DINNER + h * HEADDIM + p] = y + Dh * xh;
        __syncthreads();
    }
}

// ---------------------------------------------------------------------------
// y = y_ssm * silu(z); rmsnorm over DINNER; * norm_w; emit bf16 for GEMM2.
// One block per row.
// ---------------------------------------------------------------------------
__global__ __launch_bounds__(256)
void k_gate_norm(const float* __restrict__ z0,      // [MROWS, DINPROJ] (z = cols 0..4095)
                 const float* __restrict__ y_ssm,   // [MROWS, DINNER]
                 const float* __restrict__ norm_w,  // [DINNER]
                 unsigned short* __restrict__ ybf) {// [MROWS, DINNER] bf16
    const int row = blockIdx.x;
    const int t   = threadIdx.x;
    const float* zr = z0 + (size_t)row * DINPROJ;
    const float* yr = y_ssm + (size_t)row * DINNER;

    float v[16];
    float ss = 0.f;
#pragma unroll
    for (int i = 0; i < 16; ++i) {
        int c = t + i * 256;
        float z = zr[c];
        float s = z / (1.f + __expf(-z));
        float val = yr[c] * s;
        v[i] = val;
        ss += val * val;
    }
    __shared__ float red[256];
    red[t] = ss;
    __syncthreads();
    for (int o = 128; o > 0; o >>= 1) {
        if (t < o) red[t] += red[t + o];
        __syncthreads();
    }
    float scale = rsqrtf(red[0] * (1.f / DINNER) + 1e-5f);

    unsigned short* yo = ybf + (size_t)row * DINNER;
#pragma unroll
    for (int i = 0; i < 16; ++i) {
        int c = t + i * 256;
        yo[c] = f32_to_bf16_rne(v[i] * scale * norm_w[c]);
    }
}

// ---------------------------------------------------------------------------
// Classifier head: out[row, c] = dot(h[row,:], cls_w[c,:]) + cls_b[c]
// One block per row; 4 dots reduced in LDS.
// ---------------------------------------------------------------------------
__global__ __launch_bounds__(256)
void k_cls(const float* __restrict__ h,       // [MROWS, DMODEL]
           const float* __restrict__ cls_w,   // [NCLS, DMODEL]
           const float* __restrict__ cls_b,   // [NCLS]
           float* __restrict__ out) {         // [MROWS, NCLS]
    const int row = blockIdx.x;
    const int t   = threadIdx.x;
    const float* hr = h + (size_t)row * DMODEL;
    float p0 = 0.f, p1 = 0.f, p2 = 0.f, p3 = 0.f;
    for (int c = t; c < DMODEL; c += 256) {
        float x = hr[c];
        p0 += x * cls_w[c];
        p1 += x * cls_w[DMODEL + c];
        p2 += x * cls_w[2 * DMODEL + c];
        p3 += x * cls_w[3 * DMODEL + c];
    }
    __shared__ float red[4][256];
    red[0][t] = p0; red[1][t] = p1; red[2][t] = p2; red[3][t] = p3;
    __syncthreads();
    for (int o = 128; o > 0; o >>= 1) {
        if (t < o) {
#pragma unroll
            for (int j = 0; j < 4; ++j) red[j][t] += red[j][t + o];
        }
        __syncthreads();
    }
    if (t < NCLS) out[(size_t)row * NCLS + t] = red[t][0] + cls_b[t];
}

// ---------------------------------------------------------------------------
// Host-side launcher
// ---------------------------------------------------------------------------
extern "C" void kernel_launch(void* const* d_in, const int* in_sizes, int n_in,
                              void* d_out, int out_size, void* d_ws, size_t ws_size,
                              hipStream_t stream) {
    (void)in_sizes; (void)n_in; (void)out_size; (void)ws_size;

    const float* x         = (const float*)d_in[0];   // [B,L,DMODEL]
    const float* in_proj_w = (const float*)d_in[1];   // [DINPROJ, DMODEL]
    const float* conv_w    = (const float*)d_in[2];   // [CONVDIM, 4]
    const float* conv_b    = (const float*)d_in[3];   // [CONVDIM]
    const float* dt_bias   = (const float*)d_in[4];   // [NHEADS]
    const float* A_log     = (const float*)d_in[5];   // [NHEADS]
    const float* Dp        = (const float*)d_in[6];   // [NHEADS]
    const float* norm_w    = (const float*)d_in[7];   // [DINNER]
    const float* out_proj  = (const float*)d_in[8];   // [DMODEL, DINNER]
    const float* cls_w     = (const float*)d_in[9];   // [NCLS, DMODEL]
    const float* cls_b     = (const float*)d_in[10];  // [NCLS]
    float* out = (float*)d_out;

    // ---- workspace carve-up (256B aligned) ----
    char* ws = (char*)d_ws;
    size_t off = 0;
    auto carve = [&](size_t bytes) -> char* {
        char* p = ws + off;
        off += (bytes + 255) & ~(size_t)255;
        return p;
    };
    unsigned short* xb   = (unsigned short*)carve((size_t)MROWS * DMODEL * 2);
    unsigned short* w1b  = (unsigned short*)carve((size_t)DINPROJ * DMODEL * 2);
    unsigned short* w2b  = (unsigned short*)carve((size_t)DMODEL * DINNER * 2);
    float*          z0   = (float*)carve((size_t)MROWS * DINPROJ * 4);
    float*          xbc  = (float*)carve((size_t)MROWS * CONVDIM * 4);
    float*          dtv  = (float*)carve((size_t)MROWS * NHEADS * 4);
    float*          dAv  = (float*)carve((size_t)MROWS * NHEADS * 4);
    float*          yssm = (float*)carve((size_t)MROWS * DINNER * 4);
    unsigned short* ybf  = (unsigned short*)carve((size_t)MROWS * DINNER * 2);
    float*          hout = (float*)carve((size_t)MROWS * DMODEL * 4);

    // 1) bf16 operand conversion
    {
        size_t n0 = (size_t)MROWS * DMODEL;
        size_t n1 = (size_t)DINPROJ * DMODEL;
        size_t n2 = (size_t)DMODEL * DINNER;
        k_f32_to_bf16<<<(unsigned)((n0 + 255) / 256), 256, 0, stream>>>(x, xb, n0);
        k_f32_to_bf16<<<(unsigned)((n1 + 255) / 256), 256, 0, stream>>>(in_proj_w, w1b, n1);
        k_f32_to_bf16<<<(unsigned)((n2 + 255) / 256), 256, 0, stream>>>(out_proj, w2b, n2);
    }

    // 2) in_proj GEMM: z0[8192, 8384] = xb * w1b^T   (K = 2048)
    k_wmma_gemm_bf16<<<dim3(DINPROJ / 64, MROWS / 512), 256, 0, stream>>>(
        xb, w1b, z0, MROWS, DINPROJ, DMODEL);

    // 3) depthwise conv + SiLU
    {
        size_t total = (size_t)MROWS * CONVDIM;
        k_conv_silu<<<(unsigned)((total + 255) / 256), 256, 0, stream>>>(
            z0, conv_w, conv_b, xbc);
    }

    // 4) dt / dA
    k_dt<<<(MROWS * NHEADS + 255) / 256, 256, 0, stream>>>(z0, dt_bias, A_log, dtv, dAv);

    // 5) sequential SSM scan
    k_ssm_scan<<<dim3(NHEADS, BSZ), 64, 0, stream>>>(xbc, dtv, dAv, Dp, yssm);

    // 6) gating + RMSNorm -> bf16
    k_gate_norm<<<MROWS, 256, 0, stream>>>(z0, yssm, norm_w, ybf);

    // 7) out_proj GEMM: hout[8192, 2048] = ybf * w2b^T   (K = 4096)
    k_wmma_gemm_bf16<<<dim3(DMODEL / 64, MROWS / 512), 256, 0, stream>>>(
        ybf, w2b, hout, MROWS, DMODEL, DINNER);

    // 8) classifier head
    k_cls<<<MROWS, 256, 0, stream>>>(hout, cls_w, cls_b, out);
}
